// CoMemoryAttentionModule_55473797595844
// MI455X (gfx1250) — matrix operand order repaired and checked by reading, not compiled, not run on hardware
//
#include <hip/hip_runtime.h>

#define HEADS 8
#define KD    32
#define NB    2       // batch
#define SQ    1024
#define SK    5120
#define CH    256
#define TK    5

typedef __attribute__((ext_vector_type(16))) _Float16 v16h;
typedef __attribute__((ext_vector_type(8)))  _Float16 v8h;
typedef __attribute__((ext_vector_type(8)))  float    v8f;
typedef __attribute__((ext_vector_type(4)))  int      v4i;

#if __has_builtin(__builtin_amdgcn_global_load_async_to_lds_b128) && \
    __has_builtin(__builtin_amdgcn_s_wait_asynccnt)
#define HAS_ASYNC_LDS 1
#else
#define HAS_ASYNC_LDS 0
#endif

static __device__ __forceinline__ void async_cp16(void* lds, const void* g) {
#if HAS_ASYNC_LDS
  __builtin_amdgcn_global_load_async_to_lds_b128(
      (__attribute__((address_space(1))) v4i*)(g),
      (__attribute__((address_space(3))) v4i*)(lds), 0, 0);
#else
  *(v8h*)lds = *(const v8h*)g;     // synchronous fallback
#endif
}
static __device__ __forceinline__ void async_wait_le8() {
#if HAS_ASYNC_LDS
  __builtin_amdgcn_s_wait_asynccnt(8);
#endif
}
static __device__ __forceinline__ void async_wait_0() {
#if HAS_ASYNC_LDS
  __builtin_amdgcn_s_wait_asynccnt(0);
#endif
}

static __device__ __forceinline__ v16h cat8(v8h lo, v8h hi) {
  return __builtin_shufflevector(lo, hi, 0,1,2,3,4,5,6,7,8,9,10,11,12,13,14,15);
}
static __device__ __forceinline__ v8f wmma_f16(v16h a, v16h b, v8f c) {
  // D = A(16x32 f16) * B(32x16 f16) + C(16x16 f32)
  return __builtin_amdgcn_wmma_f32_16x16x32_f16(false, a, false, b, (short)0, c, false, false);
}

// ---------------------------------------------------------------------------
// Kernel 1: fused LayerNorm for q (enc) and k/v (memory) rows -> f16
// ---------------------------------------------------------------------------
__global__ __launch_bounds__(256) void ln_all_kernel(
    const float* __restrict__ enc, const float* __restrict__ mk,
    const float* __restrict__ mv,
    const float* __restrict__ gq, const float* __restrict__ bq,
    const float* __restrict__ gm, const float* __restrict__ bm,
    _Float16* __restrict__ qln, _Float16* __restrict__ kln,
    _Float16* __restrict__ vln) {
  const int lane = threadIdx.x & 31;
  const int wv   = threadIdx.x >> 5;
  const int row  = blockIdx.x * 8 + wv;
  const int RQ = NB * SQ, RK = NB * SK;

  const float* src; const float* gam; const float* bet; _Float16* dst; int r;
  if (row < RQ)           { src = enc; gam = gq; bet = bq; dst = qln; r = row; }
  else if (row < RQ + RK) { src = mk;  gam = gm; bet = bm; dst = kln; r = row - RQ; }
  else                    { src = mv;  gam = gm; bet = bm; dst = vln; r = row - RQ - RK; }

  const float* p = src + (size_t)r * CH + lane * 8;
  float x[8];
  float4 f0 = *(const float4*)p;
  float4 f1 = *(const float4*)(p + 4);
  x[0]=f0.x; x[1]=f0.y; x[2]=f0.z; x[3]=f0.w;
  x[4]=f1.x; x[5]=f1.y; x[6]=f1.z; x[7]=f1.w;

  float s = 0.f, ss = 0.f;
#pragma unroll
  for (int i = 0; i < 8; ++i) { s += x[i]; ss += x[i] * x[i]; }
#pragma unroll
  for (int m = 1; m < 32; m <<= 1) {
    s  += __shfl_xor(s,  m, 32);
    ss += __shfl_xor(ss, m, 32);
  }
  const float mu  = s * (1.0f / 256.0f);
  const float var = ss * (1.0f / 256.0f) - mu * mu;
  const float rst = rsqrtf(var + 1e-3f);

  _Float16* q = dst + (size_t)r * CH + lane * 8;
#pragma unroll
  for (int i = 0; i < 8; ++i) {
    const int c = lane * 8 + i;
    q[i] = (_Float16)((x[i] - mu) * rst * gam[c] + bet[c]);
  }
}

// ---------------------------------------------------------------------------
// Kernel 2: weights f32 -> f16, transposed to N-major (Wt[n][k])
// ---------------------------------------------------------------------------
__global__ __launch_bounds__(256) void wcvt_kernel(
    const float* __restrict__ Wq, const float* __restrict__ Wk,
    const float* __restrict__ Wv, const float* __restrict__ Wo,
    _Float16* __restrict__ WqT, _Float16* __restrict__ WkT,
    _Float16* __restrict__ WvT, _Float16* __restrict__ WoT) {
  const int idx = blockIdx.x * 256 + threadIdx.x;   // 0..65535
  const int k = idx >> 8, n = idx & 255;
  const float* src;
  _Float16* dst;
  switch (blockIdx.y) {
    case 0: src = Wq; dst = WqT; break;
    case 1: src = Wk; dst = WkT; break;
    case 2: src = Wv; dst = WvT; break;
    default: src = Wo; dst = WoT; break;
  }
  dst[n * 256 + k] = (_Float16)src[idx];
}

// ---------------------------------------------------------------------------
// Kernel 3: projection GEMM  Y[rows x 256] = X[rows x 256] * W (via Wt).
// storeT==1: store transposed per batch as Y[b][n][s] (for V).
// ---------------------------------------------------------------------------
__global__ __launch_bounds__(256) void proj_kernel(
    const _Float16* __restrict__ X, const _Float16* __restrict__ Wt,
    _Float16* __restrict__ Y, int storeT) {
  const int lane = threadIdx.x & 31;
  const int wv   = threadIdx.x >> 5;
  const int hi = lane >> 4, ln = lane & 15;
  const int tile = blockIdx.x;
  const int n0 = wv * 16, n1 = n0 + 128;

  const _Float16* xr  = X  + (size_t)(tile * 16 + ln) * CH;
  const _Float16* b0r = Wt + (size_t)(n0 + ln) * CH;
  const _Float16* b1r = Wt + (size_t)(n1 + ln) * CH;

  v8f acc0 = {}; v8f acc1 = {};
#pragma unroll
  for (int k0 = 0; k0 < CH; k0 += 32) {
    v16h a  = cat8(*(const v8h*)(xr + k0 + hi * 16),
                   *(const v8h*)(xr + k0 + hi * 16 + 8));
    v16h b0 = *(const v16h*)(b0r + k0 + hi * 16);
    v16h b1 = *(const v16h*)(b1r + k0 + hi * 16);
    acc0 = wmma_f16(a, b0, acc0);
    acc1 = wmma_f16(a, b1, acc1);
  }

  if (!storeT) {
    _Float16* y = Y + (size_t)(tile * 16) * CH;
#pragma unroll
    for (int r = 0; r < 8; ++r) {
      const int m = r + hi * 8;
      y[(size_t)m * CH + n0 + ln] = (_Float16)acc0[r];
      y[(size_t)m * CH + n1 + ln] = (_Float16)acc1[r];
    }
  } else {
#pragma unroll
    for (int r = 0; r < 8; ++r) {
      const int grow = tile * 16 + r + hi * 8;     // row in (NB*SK)
      const int bb = grow / SK, s = grow % SK;
      _Float16* yb = Y + (size_t)bb * 256 * SK;
      yb[(size_t)(n0 + ln) * SK + s] = (_Float16)acc0[r];
      yb[(size_t)(n1 + ln) * SK + s] = (_Float16)acc1[r];
    }
  }
}

// ---------------------------------------------------------------------------
// Kernel 4: flash attention with async double-buffered K/V staging in LDS.
// Block = 128 threads (4 waves) = (b, 16-query tile, half of the heads).
// ---------------------------------------------------------------------------
__global__ __launch_bounds__(128) void attn_kernel(
    const _Float16* __restrict__ qh, const _Float16* __restrict__ kh,
    const _Float16* __restrict__ vhT, const int* __restrict__ mask,
    _Float16* __restrict__ ctx) {
  __shared__ _Float16 Kl[2][32][128];   // [buf][key][ch within half]  8KB x2
  __shared__ _Float16 Vl[2][128][32];   // [buf][ch within half][key]  8KB x2
  __shared__ _Float16 Pl[4][16][32];    // per-wave P slab              4KB

  const int tid  = threadIdx.x;
  const int lane = tid & 31;
  const int wv   = tid >> 5;            // 0..3
  const int hi = lane >> 4, ln = lane & 15;

  const int bid = blockIdx.x;           // 0..255
  const int b   = bid >> 7;
  const int hh  = (bid >> 6) & 1;       // which half of the heads
  const int qt  = bid & 63;
  const int h   = hh * 4 + wv;          // global head of this wave
  const int cb  = hh * 128;             // channel base of this block

  const _Float16* kb = kh  + (size_t)b * SK * CH;
  const _Float16* vb = vhT + (size_t)b * 256 * SK;

  // Q fragment (A, 16x32, K = full head dim)
  const _Float16* qr = qh + (size_t)(b * SQ + qt * 16 + ln) * CH + h * KD;
  const v16h aq = cat8(*(const v8h*)(qr + hi * 16),
                       *(const v8h*)(qr + hi * 16 + 8));

  // cooperative async staging of one 32-key chunk into LDS buffer `bf`
  auto issue_chunk = [&](int bf, int kc) {
#pragma unroll
    for (int i = 0; i < 4; ++i) {                       // K: 512 x 16B segs
      const int seg = tid + i * 128;
      const int key = seg >> 4, o8 = (seg & 15) * 8;
      async_cp16(&Kl[bf][key][o8], kb + (size_t)(kc + key) * CH + cb + o8);
    }
#pragma unroll
    for (int i = 0; i < 4; ++i) {                       // V: 512 x 16B segs
      const int seg = tid + i * 128;
      const int d = seg >> 2, o8 = (seg & 3) * 8;
      async_cp16(&Vl[bf][d][o8], vb + (size_t)(cb + d) * SK + kc + o8);
    }
  };

  float mrun[8], lrun[8];
#pragma unroll
  for (int r = 0; r < 8; ++r) { mrun[r] = -1e30f; lrun[r] = 0.f; }
  v8f o0 = {}; v8f o1 = {};
  const float scl = 0.17677669529663687f;   // 1/sqrt(32)
  const int NCH = SK / 32;                  // 160 chunks

  issue_chunk(0, 0);
  for (int c = 0; c < NCH; ++c) {
    const int kc = c * 32;
    const int bf = c & 1;
    if (c + 1 < NCH) {                       // prefetch next chunk, then make
      issue_chunk((c + 1) & 1, kc + 32);     // sure chunk c (first 8/thread)
      async_wait_le8();                      // has landed (ASYNCcnt in-order)
    } else {
      async_wait_0();
    }
    __syncthreads();

    const float bias = mask[b * TK + (kc >> 10)] ? 0.0f : -1e9f;

    const v16h bk0 = *(const v16h*)&Kl[bf][ln][wv * KD + hi * 16];
    const v16h bk1 = *(const v16h*)&Kl[bf][16 + ln][wv * KD + hi * 16];
    const v8f z = {};
    v8f s0 = wmma_f16(aq, bk0, z);
    v8f s1 = wmma_f16(aq, bk1, z);

    float p0[8], p1[8];
#pragma unroll
    for (int r = 0; r < 8; ++r) {
      const float a0 = s0[r] * scl + bias;
      const float a1 = s1[r] * scl + bias;
      float mx = fmaxf(a0, a1);
      mx = fmaxf(mx, __shfl_xor(mx, 1, 32));
      mx = fmaxf(mx, __shfl_xor(mx, 2, 32));
      mx = fmaxf(mx, __shfl_xor(mx, 4, 32));
      mx = fmaxf(mx, __shfl_xor(mx, 8, 32));          // row max in 16-lane group
      const float mnew = fmaxf(mrun[r], mx);
      const float esc  = __expf(mrun[r] - mnew);
      const float e0 = __expf(a0 - mnew);
      const float e1 = __expf(a1 - mnew);
      float rs = e0 + e1;
      rs += __shfl_xor(rs, 1, 32);
      rs += __shfl_xor(rs, 2, 32);
      rs += __shfl_xor(rs, 4, 32);
      rs += __shfl_xor(rs, 8, 32);                    // row sum
      lrun[r] = lrun[r] * esc + rs;
      mrun[r] = mnew;
      o0[r] *= esc; o1[r] *= esc;
      p0[r] = e0; p1[r] = e1;
    }

    // C-layout f32 -> row-major f16 P in LDS (per-wave slab)
#pragma unroll
    for (int r = 0; r < 8; ++r) {
      const int m = r + hi * 8;
      Pl[wv][m][ln]      = (_Float16)p0[r];
      Pl[wv][m][16 + ln] = (_Float16)p1[r];
    }
    const v16h ap = cat8(*(const v8h*)&Pl[wv][ln][hi * 16],
                         *(const v8h*)&Pl[wv][ln][hi * 16 + 8]);

    const v16h bv0 = *(const v16h*)&Vl[bf][wv * KD + ln][hi * 16];
    const v16h bv1 = *(const v16h*)&Vl[bf][wv * KD + 16 + ln][hi * 16];
    o0 = wmma_f16(ap, bv0, o0);
    o1 = wmma_f16(ap, bv1, o1);

    __syncthreads();                         // buffer reuse fence
  }

  _Float16* cr = ctx + (size_t)(b * SQ + qt * 16) * CH + h * KD;
#pragma unroll
  for (int r = 0; r < 8; ++r) {
    const int m = r + hi * 8;
    const float inv = 1.0f / lrun[r];
    cr[(size_t)m * CH + ln]      = (_Float16)(o0[r] * inv);
    cr[(size_t)m * CH + 16 + ln] = (_Float16)(o1[r] * inv);
  }
}

// ---------------------------------------------------------------------------
// Kernel 5: output GEMM  out[2048 x 256] = ctx * Wo  (f32 store)
// ---------------------------------------------------------------------------
__global__ __launch_bounds__(256) void outgemm_kernel(
    const _Float16* __restrict__ X, const _Float16* __restrict__ Wt,
    float* __restrict__ out) {
  const int lane = threadIdx.x & 31;
  const int wv   = threadIdx.x >> 5;
  const int hi = lane >> 4, ln = lane & 15;
  const int tile = blockIdx.x;
  const int n0 = wv * 16, n1 = n0 + 128;

  const _Float16* xr  = X  + (size_t)(tile * 16 + ln) * CH;
  const _Float16* b0r = Wt + (size_t)(n0 + ln) * CH;
  const _Float16* b1r = Wt + (size_t)(n1 + ln) * CH;

  v8f acc0 = {}; v8f acc1 = {};
#pragma unroll
  for (int k0 = 0; k0 < CH; k0 += 32) {
    v16h a  = cat8(*(const v8h*)(xr + k0 + hi * 16),
                   *(const v8h*)(xr + k0 + hi * 16 + 8));
    v16h b0 = *(const v16h*)(b0r + k0 + hi * 16);
    v16h b1 = *(const v16h*)(b1r + k0 + hi * 16);
    acc0 = wmma_f16(a, b0, acc0);
    acc1 = wmma_f16(a, b1, acc1);
  }

  float* y = out + (size_t)(tile * 16) * CH;
#pragma unroll
  for (int r = 0; r < 8; ++r) {
    const int m = r + hi * 8;
    y[(size_t)m * CH + n0 + ln] = acc0[r];
    y[(size_t)m * CH + n1 + ln] = acc1[r];
  }
}

// ---------------------------------------------------------------------------
extern "C" void kernel_launch(void* const* d_in, const int* in_sizes, int n_in,
                              void* d_out, int out_size, void* d_ws, size_t ws_size,
                              hipStream_t stream) {
  (void)in_sizes; (void)n_in; (void)out_size; (void)ws_size;
  const float* enc = (const float*)d_in[0];
  const float* mk  = (const float*)d_in[1];
  const float* mv  = (const float*)d_in[2];
  const float* Wq  = (const float*)d_in[3];
  const float* Wk  = (const float*)d_in[4];
  const float* Wv  = (const float*)d_in[5];
  const float* Wo  = (const float*)d_in[6];
  const float* gq  = (const float*)d_in[7];
  const float* bq  = (const float*)d_in[8];
  const float* gm  = (const float*)d_in[9];
  const float* bm  = (const float*)d_in[10];
  const int*   msk = (const int*)d_in[11];

  char* p = (char*)d_ws;
  auto alloc = [&](size_t nhalves) -> _Float16* {
    _Float16* r = (_Float16*)p;
    p += ((nhalves * sizeof(_Float16) + 255) / 256) * 256;
    return r;
  };
  _Float16* qln = alloc((size_t)NB * SQ * CH);
  _Float16* kln = alloc((size_t)NB * SK * CH);
  _Float16* vln = alloc((size_t)NB * SK * CH);
  _Float16* WqT = alloc(256 * 256);
  _Float16* WkT = alloc(256 * 256);
  _Float16* WvT = alloc(256 * 256);
  _Float16* WoT = alloc(256 * 256);
  _Float16* qhB = alloc((size_t)NB * SQ * CH);
  _Float16* khB = alloc((size_t)NB * SK * CH);
  _Float16* vhT = alloc((size_t)NB * 256 * SK);
  _Float16* ctx = alloc((size_t)NB * SQ * CH);

  // 1) LayerNorms (22528 rows, 8 rows / block)
  ln_all_kernel<<<2816, 256, 0, stream>>>(enc, mk, mv, gq, bq, gm, bm,
                                          qln, kln, vln);
  // 2) weights -> transposed f16
  wcvt_kernel<<<dim3(256, 4), 256, 0, stream>>>(Wq, Wk, Wv, Wo,
                                                WqT, WkT, WvT, WoT);
  // 3) projections
  proj_kernel<<<(NB * SQ) / 16, 256, 0, stream>>>(qln, WqT, qhB, 0);
  proj_kernel<<<(NB * SK) / 16, 256, 0, stream>>>(kln, WkT, khB, 0);
  proj_kernel<<<(NB * SK) / 16, 256, 0, stream>>>(vln, WvT, vhT, 1);
  // 4) attention: 256 blocks x 128 threads (4 waves = 4 heads each)
  attn_kernel<<<256, 128, 0, stream>>>(qhB, khB, vhT, msk, ctx);
  // 5) output projection -> f32
  outgemm_kernel<<<(NB * SQ) / 16, 256, 0, stream>>>(ctx, WoT, (float*)d_out);
}
